// GAT_65386582114682
// MI455X (gfx1250) — matrix operand order repaired and checked by reading, not compiled, not run on hardware
//
#include <hip/hip_runtime.h>
#include <math.h>

// Problem constants (match reference)
#define NNODES 10000
#define NEDGES 320000
#define ETOT   (NEDGES + NNODES)   // edges + self loops
#define INF_   256
#define FEAT   512                 // H*HID == H*OUT
#define HEADS  8
#define CH     64
#define NEG_SLOPE 0.2f

typedef __attribute__((ext_vector_type(16))) _Float16 v16h;
typedef __attribute__((ext_vector_type(8)))  _Float16 v8h;
typedef __attribute__((ext_vector_type(8)))  float    v8f;

// ---------------- small helpers ----------------
__device__ __forceinline__ float leaky(float a) { return a > 0.0f ? a : NEG_SLOPE * a; }

// order-preserving float -> uint key for atomic max (works for negatives)
__device__ __forceinline__ unsigned fkey(float f) {
    unsigned b = __float_as_uint(f);
    return (b & 0x80000000u) ? ~b : (b | 0x80000000u);
}
__device__ __forceinline__ float finv(unsigned k) {
    return (k & 0x80000000u) ? __uint_as_float(k & 0x7FFFFFFFu) : __uint_as_float(~k);
}

__device__ __forceinline__ void edge_sd(const int* __restrict__ ei, int e, int& s, int& d) {
    if (e < NEDGES) { s = ei[e]; d = ei[NEDGES + e]; }
    else            { s = d = e - NEDGES; }
}

// ---------------- conversion / packing ----------------
__global__ void k_f2h(const float* __restrict__ src, _Float16* __restrict__ dst, int n) {
    int i = blockIdx.x * blockDim.x + threadIdx.x;
    if (i < n) dst[i] = (_Float16)src[i];
}

// Pack row-major W[K][Nout] (f32) into WMMA B fragments (f16).
// Fragment fi = nt*KT + kt holds 32 lanes x 16 halves:
//   lane l, half i  ->  K = kt*32 + (i&7) + (i>=8 ? 16 : 0) + (l>=16 ? 8 : 0),  n = nt*16 + (l&15)
__global__ void k_pack_B(const float* __restrict__ W, _Float16* __restrict__ Bp,
                         int K, int Nout) {
    int idx = blockIdx.x * blockDim.x + threadIdx.x;
    int total = K * Nout;
    if (idx >= total) return;
    int KT = K >> 5;
    int fi = idx >> 9;          // 512 elements per fragment
    int w  = idx & 511;
    int l  = w >> 4;
    int i  = w & 15;
    int kt = fi % KT;
    int nt = fi / KT;
    int k  = (kt << 5) + (i & 7) + ((i & 8) ? 16 : 0) + ((l & 16) ? 8 : 0);
    int n  = (nt << 4) + (l & 15);
    Bp[idx] = (_Float16)W[(size_t)k * Nout + n];
}

// ---------------- WMMA GEMM: C[M,Nout] = A[M,K](f16) * Bpacked(f16) ----------------
// One wave computes a 16(M) x 64(N) strip; K stepped by 32.
__global__ __launch_bounds__(256) void k_gemm_wmma(const _Float16* __restrict__ A,
                                                   const _Float16* __restrict__ Bp,
                                                   float* __restrict__ C,
                                                   int M, int K, int Nout) {
    int wave = (blockIdx.x * blockDim.x + threadIdx.x) >> 5;
    int lane = threadIdx.x & 31;
    int mtiles  = M >> 4;
    int ngroups = Nout >> 6;
    if (wave >= mtiles * ngroups) return;          // wave-uniform: EXEC stays all-ones
    int mtile = wave % mtiles;
    int ng    = wave / mtiles;
    int KT    = K >> 5;

    int m = (mtile << 4) + (lane & 15);
    const _Float16* arow = A + (size_t)m * K;

    v8f c0 = {}, c1 = {}, c2 = {}, c3 = {};

    for (int kt = 0; kt < KT; ++kt) {
        // A fragment (16x32 f16): lanes 0-15 hold K {0..7,16..23}, lanes 16-31 {8..15,24..31}
        int klo = (kt << 5) + ((lane & 16) ? 8 : 0);
        v8h alo = *(const v8h*)(arow + klo);
        v8h ahi = *(const v8h*)(arow + klo + 16);
        v16h a;
#pragma unroll
        for (int i = 0; i < 8; ++i) { a[i] = alo[i]; a[i + 8] = ahi[i]; }

        size_t fb = ((size_t)(ng * 4) * KT + kt) * 512 + (size_t)lane * 16;
        {
            v16h b = *(const v16h*)(Bp + fb);
            c0 = __builtin_amdgcn_wmma_f32_16x16x32_f16(false, a, false, b, (short)0, c0, false, false);
        }
        {
            v16h b = *(const v16h*)(Bp + fb + (size_t)KT * 512);
            c1 = __builtin_amdgcn_wmma_f32_16x16x32_f16(false, a, false, b, (short)0, c1, false, false);
        }
        {
            v16h b = *(const v16h*)(Bp + fb + (size_t)2 * KT * 512);
            c2 = __builtin_amdgcn_wmma_f32_16x16x32_f16(false, a, false, b, (short)0, c2, false, false);
        }
        {
            v16h b = *(const v16h*)(Bp + fb + (size_t)3 * KT * 512);
            c3 = __builtin_amdgcn_wmma_f32_16x16x32_f16(false, a, false, b, (short)0, c3, false, false);
        }
    }

    // C/D layout: VGPR r -> row M = r (lanes 0-15) / 8+r (lanes 16-31); col N = lane&15
    int colbase = (ng << 6) + (lane & 15);
    int rowbase = (mtile << 4) + ((lane & 16) ? 8 : 0);
#pragma unroll
    for (int r = 0; r < 8; ++r) {
        float* crow = C + (size_t)(rowbase + r) * Nout + colbase;
        crow[0]  = c0[r];
        crow[16] = c1[r];
        crow[32] = c2[r];
        crow[48] = c3[r];
    }
}

// ---------------- attention logits: al_src/al_dst [N,H] ----------------
__global__ void k_attn(const float* __restrict__ Hf,
                       const float* __restrict__ as_, const float* __restrict__ ad_,
                       float* __restrict__ alS, float* __restrict__ alD) {
    int t = blockIdx.x * blockDim.x + threadIdx.x;
    if (t >= NNODES * HEADS) return;
    int n = t >> 3, h = t & 7;
    const float* hp = Hf + (size_t)n * FEAT + h * CH;
    const float* sp = as_ + h * CH;
    const float* dp = ad_ + h * CH;
    float s = 0.0f, d = 0.0f;
#pragma unroll 8
    for (int c = 0; c < CH; ++c) { float v = hp[c]; s += v * sp[c]; d += v * dp[c]; }
    alS[t] = s; alD[t] = d;
}

// ---------------- edge passes ----------------
__global__ void k_edge_max(const int* __restrict__ ei, const float* __restrict__ ew,
                           const float* __restrict__ alS, const float* __restrict__ alD,
                           unsigned* __restrict__ amaxk) {
    int e = blockIdx.x * blockDim.x + threadIdx.x;
    if (e >= ETOT) return;
    int s, d; edge_sd(ei, e, s, d);
    float lew = (e < NEDGES) ? log2f(ew[e]) : 0.0f;
#pragma unroll
    for (int h = 0; h < HEADS; ++h) {
        float a = leaky(alS[s * HEADS + h] + alD[d * HEADS + h]) + lew;
        atomicMax(&amaxk[d * HEADS + h], fkey(a));
    }
}

__global__ void k_edge_sum(const int* __restrict__ ei, const float* __restrict__ ew,
                           const float* __restrict__ alS, const float* __restrict__ alD,
                           const unsigned* __restrict__ amaxk,
                           float* __restrict__ wal, float* __restrict__ denom) {
    int e = blockIdx.x * blockDim.x + threadIdx.x;
    if (e >= ETOT) return;
    int s, d; edge_sd(ei, e, s, d);
    float lew = (e < NEDGES) ? log2f(ew[e]) : 0.0f;
#pragma unroll
    for (int h = 0; h < HEADS; ++h) {
        float a  = leaky(alS[s * HEADS + h] + alD[d * HEADS + h]) + lew;
        float ex = expf(a - finv(amaxk[d * HEADS + h]));
        wal[(size_t)e * HEADS + h] = ex;
        atomicAdd(&denom[d * HEADS + h], ex);
    }
}

__global__ void k_norm(const int* __restrict__ ei, float* __restrict__ wal,
                       const float* __restrict__ denom) {
    int t = blockIdx.x * blockDim.x + threadIdx.x;
    if (t >= ETOT * HEADS) return;
    int e = t >> 3, h = t & 7;
    int d = (e < NEDGES) ? ei[NEDGES + e] : e - NEDGES;
    wal[t] = wal[t] / denom[d * HEADS + h];
}

// 64 threads per edge (one per channel), loop over heads; f32 atomics into dst rows
__global__ void k_agg(const int* __restrict__ ei, const float* __restrict__ wal,
                      const float* __restrict__ Hf, float* __restrict__ agg) {
    int t = blockIdx.x * blockDim.x + threadIdx.x;
    if (t >= ETOT * CH) return;
    int e = t >> 6, c = t & 63;
    int s, d; edge_sd(ei, e, s, d);
    const float* hs = Hf + (size_t)s * FEAT;
    const float* wp = wal + (size_t)e * HEADS;
    float* ap = agg + (size_t)d * FEAT;
#pragma unroll
    for (int h = 0; h < HEADS; ++h) {
        atomicAdd(&ap[h * CH + c], hs[h * CH + c] * wp[h]);
    }
}

// ---------------- epilogues ----------------
__global__ void k_bias_relu_f16(const float* __restrict__ agg, const float* __restrict__ b,
                                _Float16* __restrict__ out16) {
    int i = blockIdx.x * blockDim.x + threadIdx.x;
    if (i >= NNODES * FEAT) return;
    float v = agg[i] + b[i & (FEAT - 1)];
    out16[i] = (_Float16)(v > 0.0f ? v : 0.0f);
}

__global__ void k_bias_add(float* __restrict__ out, const float* __restrict__ b) {
    int i = blockIdx.x * blockDim.x + threadIdx.x;
    if (i >= NNODES * FEAT) return;
    out[i] += b[i & (FEAT - 1)];
}

// ---------------- launch ----------------
extern "C" void kernel_launch(void* const* d_in, const int* in_sizes, int n_in,
                              void* d_out, int out_size, void* d_ws, size_t ws_size,
                              hipStream_t stream) {
    const float* x   = (const float*)d_in[0];
    const int*   ei  = (const int*)  d_in[1];
    const float* ew  = (const float*)d_in[2];
    const float* W1  = (const float*)d_in[3];
    const float* as1 = (const float*)d_in[4];
    const float* ad1 = (const float*)d_in[5];
    const float* b1  = (const float*)d_in[6];
    const float* W2  = (const float*)d_in[7];
    const float* as2 = (const float*)d_in[8];
    const float* ad2 = (const float*)d_in[9];
    const float* b2  = (const float*)d_in[10];
    float* out = (float*)d_out;

    // workspace carve-up (256B aligned)
    char* p = (char*)d_ws;
    auto alloc = [&](size_t bytes) -> char* {
        char* r = p; p += (bytes + 255) & ~(size_t)255; return r;
    };
    _Float16* xh    = (_Float16*)alloc((size_t)NNODES * INF_ * 2);
    _Float16* W1p   = (_Float16*)alloc((size_t)INF_ * FEAT * 2);
    _Float16* W2p   = (_Float16*)alloc((size_t)FEAT * FEAT * 2);
    float*    hf    = (float*)   alloc((size_t)NNODES * FEAT * 4);  // GEMM out (both layers)
    float*    agg   = (float*)   alloc((size_t)NNODES * FEAT * 4);  // layer-1 aggregation
    _Float16* h16   = (_Float16*)alloc((size_t)NNODES * FEAT * 2);  // layer-2 GEMM input
    float*    alS   = (float*)   alloc((size_t)NNODES * HEADS * 4);
    float*    alD   = (float*)   alloc((size_t)NNODES * HEADS * 4);
    unsigned* amaxk = (unsigned*)alloc((size_t)NNODES * HEADS * 4);
    float*    denom = (float*)   alloc((size_t)NNODES * HEADS * 4);
    float*    wal   = (float*)   alloc((size_t)ETOT * HEADS * 4);

    dim3 blk(256);
    const int waves = (NNODES / 16) * (FEAT / 64);       // 5000 waves, 8 waves/block
    const int gemm_blocks = (waves + 7) / 8;

    // conversions / weight packing
    k_f2h   <<<dim3((NNODES * INF_ + 255) / 256), blk, 0, stream>>>(x, xh, NNODES * INF_);
    k_pack_B<<<dim3((INF_ * FEAT + 255) / 256),   blk, 0, stream>>>(W1, W1p, INF_, FEAT);
    k_pack_B<<<dim3((FEAT * FEAT + 255) / 256),   blk, 0, stream>>>(W2, W2p, FEAT, FEAT);

    // ---------- layer 1 ----------
    hipMemsetAsync(agg,   0, (size_t)NNODES * FEAT * 4,  stream);
    hipMemsetAsync(amaxk, 0, (size_t)NNODES * HEADS * 4, stream);
    hipMemsetAsync(denom, 0, (size_t)NNODES * HEADS * 4, stream);

    k_gemm_wmma<<<dim3(gemm_blocks), blk, 0, stream>>>(xh, W1p, hf, NNODES, INF_, FEAT);
    k_attn     <<<dim3((NNODES * HEADS + 255) / 256), blk, 0, stream>>>(hf, as1, ad1, alS, alD);
    k_edge_max <<<dim3((ETOT + 255) / 256), blk, 0, stream>>>(ei, ew, alS, alD, amaxk);
    k_edge_sum <<<dim3((ETOT + 255) / 256), blk, 0, stream>>>(ei, ew, alS, alD, amaxk, wal, denom);
    k_norm     <<<dim3((ETOT * HEADS + 255) / 256), blk, 0, stream>>>(ei, wal, denom);
    k_agg      <<<dim3((ETOT * CH + 255) / 256), blk, 0, stream>>>(ei, wal, hf, agg);
    k_bias_relu_f16<<<dim3((NNODES * FEAT + 255) / 256), blk, 0, stream>>>(agg, b1, h16);

    // ---------- layer 2 ----------
    hipMemsetAsync(amaxk, 0, (size_t)NNODES * HEADS * 4, stream);
    hipMemsetAsync(denom, 0, (size_t)NNODES * HEADS * 4, stream);
    hipMemsetAsync(out,   0, (size_t)NNODES * FEAT * 4,  stream);

    k_gemm_wmma<<<dim3(gemm_blocks), blk, 0, stream>>>(h16, W2p, hf, NNODES, FEAT, FEAT);
    k_attn     <<<dim3((NNODES * HEADS + 255) / 256), blk, 0, stream>>>(hf, as2, ad2, alS, alD);
    k_edge_max <<<dim3((ETOT + 255) / 256), blk, 0, stream>>>(ei, ew, alS, alD, amaxk);
    k_edge_sum <<<dim3((ETOT + 255) / 256), blk, 0, stream>>>(ei, ew, alS, alD, amaxk, wal, denom);
    k_norm     <<<dim3((ETOT * HEADS + 255) / 256), blk, 0, stream>>>(ei, wal, denom);
    k_agg      <<<dim3((ETOT * CH + 255) / 256), blk, 0, stream>>>(ei, wal, hf, out);
    k_bias_add <<<dim3((NNODES * FEAT + 255) / 256), blk, 0, stream>>>(out, b2);
}
